// EdgeModel_5643587027539
// MI455X (gfx1250) — matrix operand order repaired and checked by reading, not compiled, not run on hardware
//
#include <hip/hip_runtime.h>
#include <hip/hip_bf16.h>

// ---------------------------------------------------------------------------
// Fused EdgeModel MLP for MI455X (gfx1250, wave32, WMMA).
//   x[E,512] = concat(src,dest,edge_attr,u[batch])
//   h1 = lrelu(x@W1+b1) ; h2 = lrelu(h1@W2+b2) ; h3 = lrelu(h2@W3+b3)
//   out = LayerNorm(h3)*gamma + beta
//
// * One-time pack kernels convert W1/W2/W3 fp32 -> bf16 into d_ws, transposed
//   to [N][K]: a WMMA B-fragment = two contiguous global_load_b128 per lane.
//   448 KB of bf16 weights stay L2-resident, streamed by all blocks.
// * Main kernel: 256 thr = 8 wave32, 32 edges/block, activations in a 64 KB
//   LDS arena. Each wave owns a UNIQUE column group and multiplies both
//   16-row A tiles against each B fragment -> each weight byte is loaded
//   exactly once per block. K-loop has no barriers.
// ---------------------------------------------------------------------------

typedef __attribute__((ext_vector_type(16))) __bf16 v16bf;
typedef __attribute__((ext_vector_type(8)))  __bf16 v8bf;
typedef __attribute__((ext_vector_type(8)))  float  v8f;

#define NEG_SLOPE 0.01f
#define LN_EPS    1e-5f

// fp32 -> bf16 (round-to-nearest-even) as raw u16
static __device__ __forceinline__ unsigned short f2bf(float f) {
  union { float f; unsigned u; } v; v.f = f;
  unsigned r = v.u + 0x7FFFu + ((v.u >> 16) & 1u);
  return (unsigned short)(r >> 16);
}

static __device__ __forceinline__ v16bf cat8(v8bf lo, v8bf hi) {
  return __builtin_shufflevector(lo, hi, 0,1,2,3,4,5,6,7,8,9,10,11,12,13,14,15);
}

// ---------------------------------------------------------------------------
// Weight pack: out[n*K + k] = bf16(W[k*N + n])   (transpose to B-frag layout)
// ---------------------------------------------------------------------------
__global__ __launch_bounds__(256)
void pack_weights_kernel(const float* __restrict__ W,
                         unsigned short* __restrict__ out, int K, int N) {
  const int id = blockIdx.x * 256 + threadIdx.x;   // linear over input (coalesced read)
  if (id >= K * N) return;
  const int k = id / N;
  const int n = id - k * N;
  out[n * K + k] = f2bf(W[id]);
}

// ---------------------------------------------------------------------------
// One MLP layer on a 32-row bf16 LDS tile.
// Wave w owns column tiles [w*NT, (w+1)*NT) and both 16-row tiles, so every
// B fragment is fetched once per block and reused for two WMMAs.
// ---------------------------------------------------------------------------
template <int K, int N, bool F32OUT, bool PACKED>
static __device__ __forceinline__ void mlp_layer(
    const unsigned short* __restrict__ inb,     // bf16 LDS [32 x K]
    const float* __restrict__ Wg,               // fp32 global [K x N] (fallback)
    const unsigned short* __restrict__ Wp,      // bf16 global [N x K] (packed)
    const float* __restrict__ bias,
    unsigned short* __restrict__ wbuf,          // LDS [N x 32] (fallback only)
    unsigned short* __restrict__ outb,          // bf16 LDS [32 x N]
    float* __restrict__ outf,                   // fp32 LDS [32 x N]
    int tid) {

  constexpr int NT      = (N / 16) / 8;   // col-tiles per wave (8 waves split cols)
  constexpr int NCHUNKS = K / 32;

  const int lane = tid & 31;
  const int w    = tid >> 5;              // wave id = column group
  const int l15  = lane & 15;
  const int hiL  = lane >> 4;

  v8f acc[2][NT] = {};

  // A-fragment bases: row tiles 0 (rows 0-15) and 1 (rows 16-31)
  // layout: lanes 0-15 K={0..7,16..23}, lanes 16-31 K={8..15,24..31}
  const unsigned short* a0base = inb + (l15)      * K + hiL * 8;
  const unsigned short* a1base = inb + (16 + l15) * K + hiL * 8;

  if constexpr (PACKED) {
    const unsigned short* bbase[NT];
    #pragma unroll
    for (int j = 0; j < NT; ++j)
      bbase[j] = Wp + ((w * NT + j) * 16 + l15) * K + hiL * 8;

    #pragma unroll 2
    for (int k0 = 0; k0 < NCHUNKS; ++k0) {
      if (k0 + 1 < NCHUNKS)   // near-cache prefetch of next weight chunk
        __builtin_prefetch(bbase[0] + (k0 + 1) * 32, 0, 3);
      const v16bf a0 = cat8(*(const v8bf*)(a0base + k0 * 32),
                            *(const v8bf*)(a0base + k0 * 32 + 16));
      const v16bf a1 = cat8(*(const v8bf*)(a1base + k0 * 32),
                            *(const v8bf*)(a1base + k0 * 32 + 16));
      #pragma unroll
      for (int j = 0; j < NT; ++j) {
        const v16bf b = cat8(*(const v8bf*)(bbase[j] + k0 * 32),
                             *(const v8bf*)(bbase[j] + k0 * 32 + 16));
        acc[0][j] = __builtin_amdgcn_wmma_f32_16x16x32_bf16(
            false, a0, false, b, (short)0, acc[0][j], false, false);
        acc[1][j] = __builtin_amdgcn_wmma_f32_16x16x32_bf16(
            false, a1, false, b, (short)0, acc[1][j], false, false);
      }
    }
  } else {
    // -------- fallback: stage fp32 weights through LDS --------
    for (int k0 = 0; k0 < NCHUNKS; ++k0) {
      const float* Wc = Wg + (k0 * 32) * N;
      for (int id = tid; id < 32 * N; id += 256) {
        const int k = id / N;
        const int n = id - k * N;
        wbuf[n * 32 + k] = f2bf(Wc[id]);
      }
      __syncthreads();
      const v16bf a0 = cat8(*(const v8bf*)(a0base + k0 * 32),
                            *(const v8bf*)(a0base + k0 * 32 + 16));
      const v16bf a1 = cat8(*(const v8bf*)(a1base + k0 * 32),
                            *(const v8bf*)(a1base + k0 * 32 + 16));
      #pragma unroll
      for (int j = 0; j < NT; ++j) {
        const unsigned short* bp = wbuf + ((w * NT + j) * 16 + l15) * 32 + hiL * 8;
        const v16bf b = cat8(*(const v8bf*)(bp), *(const v8bf*)(bp + 16));
        acc[0][j] = __builtin_amdgcn_wmma_f32_16x16x32_bf16(
            false, a0, false, b, (short)0, acc[0][j], false, false);
        acc[1][j] = __builtin_amdgcn_wmma_f32_16x16x32_bf16(
            false, a1, false, b, (short)0, acc[1][j], false, false);
      }
      __syncthreads();
    }
  }

  // ---- epilogue: bias + LeakyReLU; C/D layout: VGPR r -> M=r (+8 hi lanes) ----
  #pragma unroll
  for (int rt = 0; rt < 2; ++rt) {
    const int mbase = rt * 16 + hiL * 8;
    #pragma unroll
    for (int j = 0; j < NT; ++j) {
      const int col = (w * NT + j) * 16 + l15;
      const float bv = bias[col];
      #pragma unroll
      for (int r = 0; r < 8; ++r) {
        float v = acc[rt][j][r] + bv;
        v = (v >= 0.f) ? v : NEG_SLOPE * v;
        const int m = mbase + r;
        if constexpr (F32OUT) outf[m * N + col] = v;
        else                  outb[m * N + col] = f2bf(v);
      }
    }
  }
}

template <bool PACKED>
__global__ __launch_bounds__(256)
void EdgeModel_fused_wmma_kernel(
    const float* __restrict__ src, const float* __restrict__ dest,
    const float* __restrict__ edge, const float* __restrict__ u,
    const int*   __restrict__ batch,
    const float* __restrict__ W1, const float* __restrict__ b1,
    const float* __restrict__ W2, const float* __restrict__ b2,
    const float* __restrict__ W3, const float* __restrict__ b3,
    const unsigned short* __restrict__ wp1,
    const unsigned short* __restrict__ wp2,
    const unsigned short* __restrict__ wp3,
    const float* __restrict__ gamma, const float* __restrict__ beta,
    float* __restrict__ out, int E) {

  // 64 KB LDS arena:
  //   [0      , 32768) xs : 32x512 bf16  (fallback: wbuf for layers 2/3)
  //   [32768  , 49152) hA : 32x256 bf16  (later: fp32 h3 32x128)
  //   [49152  , 65536) hB : 32x256 bf16  (fallback: wbuf for layer 1)
  __shared__ __align__(16) unsigned char smem[65536];
  unsigned short* xs = (unsigned short*)(smem);
  unsigned short* hA = (unsigned short*)(smem + 32768);
  unsigned short* hB = (unsigned short*)(smem + 49152);
  float*          h3 = (float*)(smem + 32768);

  const int tid = threadIdx.x;
  const int e0  = blockIdx.x * 32;

  // ---- stage x tile: float4 loads -> packed bf16x4 (uint2) LDS stores ----
  #pragma unroll 4
  for (int q = tid; q < 32 * 512 / 4; q += 256) {
    const int r = q >> 7;            // row in tile
    const int c = (q & 127) << 2;    // starting col (quad never crosses a segment)
    int e = e0 + r; if (e >= E) e = E - 1;
    const float* p;
    if      (c < 128) p = src  + e * 128 + c;
    else if (c < 256) p = dest + e * 128 + (c - 128);
    else if (c < 384) p = edge + e * 128 + (c - 256);
    else              p = u + batch[e] * 128 + (c - 384);
    const float4 f = *(const float4*)p;
    uint2 d;
    d.x = (unsigned)f2bf(f.x) | ((unsigned)f2bf(f.y) << 16);
    d.y = (unsigned)f2bf(f.z) | ((unsigned)f2bf(f.w) << 16);
    *(uint2*)(xs + r * 512 + c) = d;
  }
  __syncthreads();

  mlp_layer<512, 256, false, PACKED>(xs, W1, wp1, b1, hB, hA, nullptr, tid);
  __syncthreads();
  mlp_layer<256, 256, false, PACKED>(hA, W2, wp2, b2, xs, hB, nullptr, tid);
  __syncthreads();
  mlp_layer<256, 128, true,  PACKED>(hB, W3, wp3, b3, xs, nullptr, h3, tid);
  __syncthreads();

  // ---- LayerNorm over 128 cols: 8 lanes/row, wave32 shfl_xor reduction ----
  const int row = tid >> 3;
  const int seg = tid & 7;
  float vals[16];
  float s = 0.f, s2 = 0.f;
  #pragma unroll
  for (int i = 0; i < 16; ++i) {
    const float v = h3[row * 128 + seg * 16 + i];
    vals[i] = v; s += v; s2 += v * v;
  }
  #pragma unroll
  for (int m = 1; m < 8; m <<= 1) {
    s  += __shfl_xor(s,  m, 32);
    s2 += __shfl_xor(s2, m, 32);
  }
  const float mean = s * (1.f / 128.f);
  const float var  = s2 * (1.f / 128.f) - mean * mean;
  const float inv  = rsqrtf(var + LN_EPS);
  const int e = e0 + row;
  if (e < E) {
    #pragma unroll
    for (int i = 0; i < 16; ++i) {
      const int c = seg * 16 + i;
      out[e * 128 + c] = (vals[i] - mean) * inv * gamma[c] + beta[c];
    }
  }
}

extern "C" void kernel_launch(void* const* d_in, const int* in_sizes, int n_in,
                              void* d_out, int out_size, void* d_ws, size_t ws_size,
                              hipStream_t stream) {
  (void)n_in; (void)out_size;
  const float* src   = (const float*)d_in[0];
  const float* dest  = (const float*)d_in[1];
  const float* edge  = (const float*)d_in[2];
  const float* u     = (const float*)d_in[3];
  const int*   batch = (const int*)  d_in[4];
  const float* W1 = (const float*)d_in[5];
  const float* b1 = (const float*)d_in[6];
  const float* W2 = (const float*)d_in[7];
  const float* b2 = (const float*)d_in[8];
  const float* W3 = (const float*)d_in[9];
  const float* b3 = (const float*)d_in[10];
  const float* gm = (const float*)d_in[11];
  const float* bt = (const float*)d_in[12];
  float* out = (float*)d_out;

  const int E = in_sizes[0] / 128;          // 400000
  const int nblocks = (E + 31) / 32;        // 12500

  // packed bf16 weight layout in d_ws (halves):
  //   wp1 [256][512] @ 0 ; wp2 [256][256] @ 131072 ; wp3 [128][256] @ 196608
  constexpr size_t PACK_HALVES = 131072 + 65536 + 32768;   // 229376
  unsigned short* wp  = (unsigned short*)d_ws;
  unsigned short* wp1 = wp;
  unsigned short* wp2 = wp + 131072;
  unsigned short* wp3 = wp + 196608;

  if (ws_size >= PACK_HALVES * sizeof(unsigned short)) {
    pack_weights_kernel<<<dim3((512 * 256 + 255) / 256), dim3(256), 0, stream>>>(W1, wp1, 512, 256);
    pack_weights_kernel<<<dim3((256 * 256 + 255) / 256), dim3(256), 0, stream>>>(W2, wp2, 256, 256);
    pack_weights_kernel<<<dim3((256 * 128 + 255) / 256), dim3(256), 0, stream>>>(W3, wp3, 256, 128);
    EdgeModel_fused_wmma_kernel<true><<<dim3(nblocks), dim3(256), 0, stream>>>(
        src, dest, edge, u, batch, W1, b1, W2, b2, W3, b3,
        wp1, wp2, wp3, gm, bt, out, E);
  } else {
    EdgeModel_fused_wmma_kernel<false><<<dim3(nblocks), dim3(256), 0, stream>>>(
        src, dest, edge, u, batch, W1, b1, W2, b2, W3, b3,
        wp1, wp2, wp3, gm, bt, out, E);
  }
}
